// GNN_12017318494477
// MI455X (gfx1250) — compile-verified, matrix operand order checked by reading, not run on hardware
//
// CDNA5 / gfx1250 GNN message-passing network, bf16 WMMA (16x16x32) pipeline.
// One wave (32 lanes) processes a 32-row tile (2 M-subtiles) through each
// 4-layer MLP with v_wmma_f32_16x16x32_bf16; each weight (B) fragment load
// feeds 2 WMMAs. A-operands are register-resident per layer; activations
// round-trip through LDS between layers (the D-layout -> A-layout transpose)
// using truncating bf16 stores that fuse into ds_store_b16_d16_hi. Edge
// features are staged into LDS with GLOBAL_LOAD_ASYNC_TO_LDS_B128 (ASYNCcnt)
// overlapping the gather-multiply.

#include <hip/hip_runtime.h>
#include <hip/hip_bf16.h>
#include <stdint.h>

#define EMB 128
#define NNODES 20000
#define NEDGES 320000

typedef __attribute__((ext_vector_type(16))) __bf16 v16bf;
typedef __attribute__((ext_vector_type(8)))  float  v8f;

union Frag {
  v16bf bf;
  uint4 q[2];
};

// RNE conversion (weights / normalized activations).
__device__ __forceinline__ unsigned short f2bf_rn(float f) {
  unsigned u = __float_as_uint(f);
  u += 0x7FFFu + ((u >> 16) & 1u);
  return (unsigned short)(u >> 16);
}
// Truncating conversion: high half of the f32, fuses into *_d16_hi stores.
__device__ __forceinline__ unsigned short f2bf_tr(float f) {
  return (unsigned short)(__float_as_uint(f) >> 16);
}
// Pack two truncated bf16 into one dword with a single v_perm_b32.
__device__ __forceinline__ unsigned pack2bf(float lo, float hi) {
  return __builtin_amdgcn_perm(__float_as_uint(hi), __float_as_uint(lo), 0x07060302u);
}
__device__ __forceinline__ float bf2f(unsigned short h) {
  return __uint_as_float(((unsigned)h) << 16);
}
__device__ __forceinline__ float prelu(float x, float a) {
  return fmaxf(x, 0.f) + a * fminf(x, 0.f);
}

// Async global->LDS 16B copy, tracked on ASYNCcnt (CDNA5 async path).
__device__ __forceinline__ void async_copy_b128(unsigned ldsAddr, const void* g) {
  asm volatile("global_load_async_to_lds_b128 %0, %1, off"
               :: "v"(ldsAddr), "v"((unsigned long long)(uintptr_t)g)
               : "memory");
}
__device__ __forceinline__ void wait_async() {
  asm volatile("s_wait_asynccnt 0" ::: "memory");
}

// Load the (MT*16)xK A-operand (K = KT*32) from an LDS tile into registers.
template <int MT, int KT>
__device__ __forceinline__ void load_A(Frag (&a)[MT][KT], const unsigned short* in, int sIn) {
  const int lane = threadIdx.x & 31;
  const int half = lane >> 4;
  const int n    = lane & 15;
#pragma unroll
  for (int mt = 0; mt < MT; ++mt)
#pragma unroll
    for (int kt = 0; kt < KT; ++kt) {
      const unsigned short* pa = in + (mt * 16 + n) * sIn + kt * 32 + half * 8;
      a[mt][kt].q[0] = *(const uint4*)(pa);
      a[mt][kt].q[1] = *(const uint4*)(pa + 16);
    }
}

// MT 16x16 output tiles for one N-tile; each B fragment feeds MT WMMAs.
template <int MT, int KT>
__device__ __forceinline__ void acc_tiles(v8f (&acc)[MT], const Frag (&a)[MT][KT],
                                          const unsigned short* __restrict__ w,
                                          int nt, float bv) {
  const int lane = threadIdx.x & 31;
  const int half = lane >> 4;
  const int n    = lane & 15;
  v8f z = {bv, bv, bv, bv, bv, bv, bv, bv};
#pragma unroll
  for (int mt = 0; mt < MT; ++mt) acc[mt] = z;
#pragma unroll
  for (int kt = 0; kt < KT; ++kt) {
    Frag b;
    const unsigned short* pb = w + (size_t)(nt * 16 + n) * (KT * 32) + kt * 32 + half * 8;
    b.q[0] = *(const uint4*)(pb);
    b.q[1] = *(const uint4*)(pb + 16);
#pragma unroll
    for (int mt = 0; mt < MT; ++mt)
      acc[mt] = __builtin_amdgcn_wmma_f32_16x16x32_bf16(false, a[mt][kt].bf, false, b.bf,
                                                        (short)0, acc[mt], false, false);
  }
}

// Full hidden layer (128 outputs) with PReLU, D written to LDS as bf16.
template <int MT, int KT>
__device__ __forceinline__ void mlp_layer(const unsigned short* in, int sIn,
                                          unsigned short* out, int sOut,
                                          const unsigned short* __restrict__ w,
                                          const float* __restrict__ bias,
                                          float alpha) {
  const int lane = threadIdx.x & 31;
  const int half = lane >> 4;
  const int n    = lane & 15;
  Frag a[MT][KT];
  load_A<MT, KT>(a, in, sIn);
  for (int nt = 0; nt < 8; ++nt) {
    float bv = bias ? bias[nt * 16 + n] : 0.f;
    v8f acc[MT];
    acc_tiles<MT, KT>(acc, a, w, nt, bv);
#pragma unroll
    for (int mt = 0; mt < MT; ++mt)
#pragma unroll
      for (int j = 0; j < 8; ++j)
        out[(mt * 16 + j + 8 * half) * sOut + nt * 16 + n] = f2bf_tr(prelu(acc[mt][j], alpha));
  }
  __syncthreads();
}

// f32 -> bf16 (RNE) padded weight conversion. grid = rowspad, block = colspad.
__global__ void cvt_pad(const float* __restrict__ src, unsigned short* __restrict__ dst,
                        int rows, int cols, int colspad) {
  int r = blockIdx.x, c = threadIdx.x;
  float v = (r < rows && c < cols) ? src[(size_t)r * cols + c] : 0.f;
  dst[(size_t)r * colspad + c] = f2bf_rn(v);
}

// Encoder: in[g*inStride + inOff + c], c < inCols (K padded to 32). 32 rows/wave.
__global__ void __launch_bounds__(32)
enc_kernel(const float* __restrict__ in, int inStride, int inOff, int inCols,
           const unsigned short* __restrict__ w,
           const float* __restrict__ b_in, const float* __restrict__ b_out,
           const float* __restrict__ a0p, const float* __restrict__ a1p,
           const float* __restrict__ a2p,
           float* __restrict__ outF, unsigned short* __restrict__ outB) {
  __shared__ __align__(16) unsigned short ldsA[32 * 40];
  __shared__ __align__(16) unsigned short ldsB[32 * 136];
  __shared__ __align__(16) unsigned short ldsC[32 * 136];
  const int lane = threadIdx.x & 31;
  const int half = lane >> 4;
  const int n    = lane & 15;
  const int tile = blockIdx.x;
  {
    const size_t g = (size_t)tile * 32 + lane;   // one row per lane
#pragma unroll
    for (int cc = 0; cc < 32; cc += 8) {
      unsigned u[4];
#pragma unroll
      for (int p = 0; p < 4; ++p) {
        int c0 = cc + 2 * p, c1 = c0 + 1;
        float f0 = (c0 < inCols) ? in[g * inStride + inOff + c0] : 0.f;
        float f1 = (c1 < inCols) ? in[g * inStride + inOff + c1] : 0.f;
        u[p] = pack2bf(f0, f1);
      }
      uint4 q; q.x = u[0]; q.y = u[1]; q.z = u[2]; q.w = u[3];
      *(uint4*)&ldsA[lane * 40 + cc] = q;
    }
  }
  __syncthreads();
  const float a0 = *a0p, a1 = *a1p, a2 = *a2p;
  mlp_layer<2, 1>(ldsA, 40,  ldsB, 136, w,                b_in,    a0);
  mlp_layer<2, 4>(ldsB, 136, ldsC, 136, w + 4096,         nullptr, a1);
  mlp_layer<2, 4>(ldsC, 136, ldsB, 136, w + 4096 + 16384, nullptr, a2);
  Frag a[2][4];
  load_A<2, 4>(a, ldsB, 136);
  for (int nt = 0; nt < 8; ++nt) {
    v8f acc[2];
    acc_tiles<2, 4>(acc, a, w + 4096 + 32768, nt, b_out[nt * 16 + n]);
#pragma unroll
    for (int mt = 0; mt < 2; ++mt)
#pragma unroll
      for (int j = 0; j < 8; ++j) {
        size_t row = (size_t)tile * 32 + mt * 16 + j + 8 * half;
        if (outF) outF[row * EMB + nt * 16 + n] = acc[mt][j];
        else      outB[row * EMB + nt * 16 + n] = f2bf_tr(acc[mt][j]);
      }
  }
}

// GraphNorm stats: per-feature column reduction -> affine A,B.
__global__ void __launch_bounds__(256)
gn_stats(const float* __restrict__ v, const float* __restrict__ wgt,
         const float* __restrict__ bia, const float* __restrict__ ms,
         float* __restrict__ AB) {
  __shared__ float sh[256], sh2[256];
  const int c = blockIdx.x, t = threadIdx.x;
  float s = 0.f, s2 = 0.f;
  for (int i = t; i < NNODES; i += 256) {
    float x = v[(size_t)i * EMB + c];
    s += x; s2 += x * x;
  }
  sh[t] = s; sh2[t] = s2;
  __syncthreads();
  for (int o = 128; o > 0; o >>= 1) {
    if (t < o) { sh[t] += sh[t + o]; sh2[t] += sh2[t + o]; }
    __syncthreads();
  }
  if (t == 0) {
    const float inv = 1.f / (float)NNODES;
    float mean = sh[0] * inv, ex2 = sh2[0] * inv;
    float m = ms[c];
    float var = ex2 - (2.f * m - m * m) * mean * mean;
    float A = wgt[c] * rsqrtf(var + 1e-5f);
    AB[c] = A;
    AB[EMB + c] = bia[c] - A * m * mean;
  }
}

__global__ void __launch_bounds__(256)
gn_apply(float* __restrict__ v, unsigned short* __restrict__ vbf,
         const float* __restrict__ AB) {
  size_t i = (size_t)blockIdx.x * 256 + threadIdx.x;
  int c = (int)(i & (EMB - 1));
  float x = AB[c] * v[i] + AB[EMB + c];
  v[i] = x;
  vbf[i] = f2bf_rn(x);
}

__global__ void __launch_bounds__(256)
zero_f(float* __restrict__ p) {
  p[(size_t)blockIdx.x * 256 + threadIdx.x] = 0.f;
}

// Message MLP over 32-edge tiles: [v_dst*v_src || ef] -> 128, scatter-add.
__global__ void __launch_bounds__(32)
msg_kernel(const unsigned short* __restrict__ vbf, const unsigned short* __restrict__ ef,
           const int* __restrict__ srcIdx, const int* __restrict__ dstIdx,
           const unsigned short* __restrict__ w,
           const float* __restrict__ b_in, const float* __restrict__ b_out,
           const float* __restrict__ a0p, const float* __restrict__ a1p,
           const float* __restrict__ a2p, float* __restrict__ agg) {
  __shared__ __align__(16) unsigned short ldsIn[32 * 264];
  __shared__ __align__(16) unsigned short ldsB[32 * 136];
  __shared__ __align__(16) unsigned short ldsC[32 * 136];
  __shared__ int ldsD[32];
  const int lane = threadIdx.x & 31;
  const int half = lane >> 4;
  const int n    = lane & 15;
  const int tile = blockIdx.x;
  const size_t eid = (size_t)tile * 32 + lane;   // one edge row per lane

  // Kick off edge-feature copy global->LDS on the async path (ASYNCcnt),
  // overlapping with the index load + gather-multiply below.
#pragma unroll
  for (int ch = 0; ch < 16; ++ch) {
    unsigned la = (unsigned)(uintptr_t)(const void*)&ldsIn[lane * 264 + 128 + ch * 8];
    async_copy_b128(la, ef + eid * EMB + ch * 8);
  }

  const int s = srcIdx[eid];
  const int d = dstIdx[eid];
  ldsD[lane] = d;
  {
#pragma unroll
    for (int ch = 0; ch < 16; ++ch) {
      uint4 qs = *(const uint4*)(vbf + (size_t)s * EMB + ch * 8);
      uint4 qd = *(const uint4*)(vbf + (size_t)d * EMB + ch * 8);
      const unsigned short* ps = (const unsigned short*)&qs;
      const unsigned short* pd = (const unsigned short*)&qd;
      unsigned u[4];
#pragma unroll
      for (int p = 0; p < 4; ++p)
        u[p] = pack2bf(bf2f(ps[2 * p]) * bf2f(pd[2 * p]),
                       bf2f(ps[2 * p + 1]) * bf2f(pd[2 * p + 1]));
      uint4 q; q.x = u[0]; q.y = u[1]; q.z = u[2]; q.w = u[3];
      *(uint4*)&ldsIn[lane * 264 + ch * 8] = q;
    }
  }
  wait_async();
  __syncthreads();
  const float a0 = *a0p, a1 = *a1p, a2 = *a2p;
  mlp_layer<2, 8>(ldsIn, 264, ldsB, 136, w,         b_in,    a0);
  mlp_layer<2, 4>(ldsB, 136,  ldsC, 136, w + 32768, nullptr, a1);
  mlp_layer<2, 4>(ldsC, 136,  ldsB, 136, w + 49152, nullptr, a2);
  Frag a[2][4];
  load_A<2, 4>(a, ldsB, 136);
  for (int nt = 0; nt < 8; ++nt) {
    v8f acc[2];
    acc_tiles<2, 4>(acc, a, w + 65536, nt, b_out[nt * 16 + n]);
#pragma unroll
    for (int mt = 0; mt < 2; ++mt)
#pragma unroll
      for (int j = 0; j < 8; ++j) {
        int node = ldsD[mt * 16 + j + 8 * half];
        unsafeAtomicAdd(&agg[(size_t)node * EMB + nt * 16 + n], acc[mt][j]);
      }
  }
}

// Node-update MLP on aggregated messages, optional residual into v.
__global__ void __launch_bounds__(32)
outmlp_kernel(const float* __restrict__ agg, const unsigned short* __restrict__ w,
              const float* __restrict__ b_in, const float* __restrict__ b_out,
              const float* __restrict__ a0p, const float* __restrict__ a1p,
              const float* __restrict__ a2p, float* __restrict__ v, int residual) {
  __shared__ __align__(16) unsigned short ldsA[32 * 136];
  __shared__ __align__(16) unsigned short ldsB[32 * 136];
  __shared__ __align__(16) unsigned short ldsC[32 * 136];
  const int lane = threadIdx.x & 31;
  const int half = lane >> 4;
  const int n    = lane & 15;
  const int tile = blockIdx.x;
  {
    const size_t g = (size_t)tile * 32 + lane;
#pragma unroll
    for (int ch = 0; ch < 16; ++ch) {
      unsigned u[4];
#pragma unroll
      for (int p = 0; p < 4; ++p)
        u[p] = pack2bf(agg[g * EMB + ch * 8 + 2 * p], agg[g * EMB + ch * 8 + 2 * p + 1]);
      uint4 q; q.x = u[0]; q.y = u[1]; q.z = u[2]; q.w = u[3];
      *(uint4*)&ldsA[lane * 136 + ch * 8] = q;
    }
  }
  __syncthreads();
  const float a0 = *a0p, a1 = *a1p, a2 = *a2p;
  mlp_layer<2, 4>(ldsA, 136, ldsB, 136, w,         b_in,    a0);
  mlp_layer<2, 4>(ldsB, 136, ldsC, 136, w + 16384, nullptr, a1);
  mlp_layer<2, 4>(ldsC, 136, ldsB, 136, w + 32768, nullptr, a2);
  Frag a[2][4];
  load_A<2, 4>(a, ldsB, 136);
  for (int nt = 0; nt < 8; ++nt) {
    v8f acc[2];
    acc_tiles<2, 4>(acc, a, w + 49152, nt, b_out[nt * 16 + n]);
#pragma unroll
    for (int mt = 0; mt < 2; ++mt)
#pragma unroll
      for (int j = 0; j < 8; ++j) {
        size_t o = ((size_t)tile * 32 + mt * 16 + j + 8 * half) * EMB + nt * 16 + n;
        float base = residual ? v[o] : 0.f;
        v[o] = base + acc[mt][j];
      }
  }
}

// Decoder 128 -> 3 (last weight padded to 16x128 with zero rows).
__global__ void __launch_bounds__(32)
dec_kernel(const float* __restrict__ v, const unsigned short* __restrict__ w,
           const float* __restrict__ b_in, const float* __restrict__ b_out,
           const float* __restrict__ a0p, const float* __restrict__ a1p,
           const float* __restrict__ a2p, float* __restrict__ out) {
  __shared__ __align__(16) unsigned short ldsA[32 * 136];
  __shared__ __align__(16) unsigned short ldsB[32 * 136];
  __shared__ __align__(16) unsigned short ldsC[32 * 136];
  const int lane = threadIdx.x & 31;
  const int half = lane >> 4;
  const int n    = lane & 15;
  const int tile = blockIdx.x;
  {
    const size_t g = (size_t)tile * 32 + lane;
#pragma unroll
    for (int ch = 0; ch < 16; ++ch) {
      unsigned u[4];
#pragma unroll
      for (int p = 0; p < 4; ++p)
        u[p] = pack2bf(v[g * EMB + ch * 8 + 2 * p], v[g * EMB + ch * 8 + 2 * p + 1]);
      uint4 q; q.x = u[0]; q.y = u[1]; q.z = u[2]; q.w = u[3];
      *(uint4*)&ldsA[lane * 136 + ch * 8] = q;
    }
  }
  __syncthreads();
  const float a0 = *a0p, a1 = *a1p, a2 = *a2p;
  mlp_layer<2, 4>(ldsA, 136, ldsB, 136, w,         b_in,    a0);
  mlp_layer<2, 4>(ldsB, 136, ldsC, 136, w + 16384, nullptr, a1);
  mlp_layer<2, 4>(ldsC, 136, ldsB, 136, w + 32768, nullptr, a2);
  Frag a[2][4];
  load_A<2, 4>(a, ldsB, 136);
  v8f acc[2];
  acc_tiles<2, 4>(acc, a, w + 49152, 0, (n < 3) ? b_out[n] : 0.f);
  if (n < 3) {
#pragma unroll
    for (int mt = 0; mt < 2; ++mt)
#pragma unroll
      for (int j = 0; j < 8; ++j)
        out[((size_t)tile * 32 + mt * 16 + j + 8 * half) * 3 + n] = acc[mt][j];
  }
}

extern "C" void kernel_launch(void* const* d_in, const int* in_sizes, int n_in,
                              void* d_out, int out_size, void* d_ws, size_t ws_size,
                              hipStream_t stream) {
  // Input flattening assumed: x, edge_attr, params (insertion order, each MLP =
  // W0..W3, b_in, b_out, a0..a2), edge_index.
  const float* x  = (const float*)d_in[0];
  const float* ea = (const float*)d_in[1];
  const int*  eix = (const int*)d_in[101];
  const int* srcI = eix;
  const int* dstI = eix + NEDGES;
  auto F = [&](int i) { return (const float*)d_in[i]; };

  // bf16 weight arena (element offsets)
  unsigned short* wb = (unsigned short*)d_ws;
  const size_t ENC_SZ = 128 * 32 + 3 * 128 * 128;   // 53248
  const size_t MSG_SZ = 128 * 256 + 3 * 128 * 128;  // 81920
  const size_t OUT_SZ = 4 * 128 * 128;              // 65536
  const size_t DEC_SZ = 3 * 128 * 128 + 16 * 128;   // 51200
  const size_t O_NODE = 0, O_EDGE = O_NODE + ENC_SZ, O_FAR = O_EDGE + ENC_SZ;
  const size_t O_MSG0 = O_FAR + ENC_SZ;
  const size_t O_OUT0 = O_MSG0 + 3 * MSG_SZ;
  const size_t O_DEC  = O_OUT0 + 3 * OUT_SZ;
  const size_t WB_BYTES = (O_DEC + DEC_SZ) * 2;     // 1,306,624 (256B aligned)

  char* base = (char*)d_ws;
  float* v   = (float*)(base + WB_BYTES);
  float* agg = (float*)(base + WB_BYTES + (size_t)NNODES * EMB * 4);
  float* AB  = (float*)(base + WB_BYTES + 2 * (size_t)NNODES * EMB * 4);
  unsigned short* vbf = (unsigned short*)((char*)AB + 1024);
  unsigned short* ebf = (unsigned short*)((char*)vbf + (size_t)NNODES * EMB * 2);
  unsigned short* fbf = (unsigned short*)((char*)ebf + (size_t)NEDGES * EMB * 2);

  auto cvt = [&](int pidx, size_t off, int rows, int cols, int colspad, int rowspad) {
    cvt_pad<<<rowspad, colspad, 0, stream>>>(F(pidx), wb + off, rows, cols, colspad);
  };
  const int encIn[3] = {2, 11, 20};
  const size_t encOff[3] = {O_NODE, O_EDGE, O_FAR};
  const int encCols[3] = {6, 4, 1};
  for (int m = 0; m < 3; ++m) {
    cvt(encIn[m] + 0, encOff[m] + 0,     128, encCols[m], 32, 128);
    cvt(encIn[m] + 1, encOff[m] + 4096,  128, 128, 128, 128);
    cvt(encIn[m] + 2, encOff[m] + 20480, 128, 128, 128, 128);
    cvt(encIn[m] + 3, encOff[m] + 36864, 128, 128, 128, 128);
  }
  for (int l = 0; l < 3; ++l) {
    int pm = 38 + l * 18;                   // mp[l].msg
    size_t om = O_MSG0 + l * MSG_SZ;
    cvt(pm + 0, om + 0,     128, 256, 256, 128);
    cvt(pm + 1, om + 32768, 128, 128, 128, 128);
    cvt(pm + 2, om + 49152, 128, 128, 128, 128);
    cvt(pm + 3, om + 65536, 128, 128, 128, 128);
    int po = pm + 9;                        // mp[l].out
    size_t oo = O_OUT0 + l * OUT_SZ;
    for (int i = 0; i < 4; ++i) cvt(po + i, oo + i * 16384, 128, 128, 128, 128);
  }
  cvt(92, O_DEC + 0,     128, 128, 128, 128);
  cvt(93, O_DEC + 16384, 128, 128, 128, 128);
  cvt(94, O_DEC + 32768, 128, 128, 128, 128);
  cvt(95, O_DEC + 49152, 3,   128, 128, 16);

  // Encoders (32 rows per workgroup)
  enc_kernel<<<NNODES / 32, 32, 0, stream>>>(x, 6, 0, 6, wb + O_NODE,
      F(6), F(7), F(8), F(9), F(10), v, nullptr);
  enc_kernel<<<NEDGES / 32, 32, 0, stream>>>(ea, 4, 0, 4, wb + O_EDGE,
      F(15), F(16), F(17), F(18), F(19), nullptr, ebf);
  enc_kernel<<<NEDGES / 32, 32, 0, stream>>>(ea, 4, 3, 1, wb + O_FAR,
      F(24), F(25), F(26), F(27), F(28), nullptr, fbf);

  // Message-passing layers
  for (int l = 0; l < 3; ++l) {
    int pg = 29 + l * 3;
    gn_stats<<<EMB, 256, 0, stream>>>(v, F(pg), F(pg + 1), F(pg + 2), AB);
    gn_apply<<<(NNODES * EMB) / 256, 256, 0, stream>>>(v, vbf, AB);
    zero_f<<<(NNODES * EMB) / 256, 256, 0, stream>>>(agg);
    int pm = 38 + l * 18;
    msg_kernel<<<NEDGES / 32, 32, 0, stream>>>(vbf, (l == 0) ? ebf : fbf, srcI, dstI,
        wb + O_MSG0 + l * MSG_SZ, F(pm + 4), F(pm + 5), F(pm + 6), F(pm + 7), F(pm + 8), agg);
    int po = pm + 9;
    outmlp_kernel<<<NNODES / 32, 32, 0, stream>>>(agg, wb + O_OUT0 + l * OUT_SZ,
        F(po + 4), F(po + 5), F(po + 6), F(po + 7), F(po + 8), v, l != 0);
  }

  dec_kernel<<<NNODES / 32, 32, 0, stream>>>(v, wb + O_DEC,
      F(96), F(97), F(98), F(99), F(100), (float*)d_out);
}
